// RegionProposalNetwork_1297080124163
// MI455X (gfx1250) — matrix operand order repaired and checked by reading, not compiled, not run on hardware
//
#include <hip/hip_runtime.h>
#include <hip/hip_bf16.h>

typedef float v8f __attribute__((ext_vector_type(8)));
typedef unsigned int v8u __attribute__((ext_vector_type(8)));
typedef __bf16 v16bf __attribute__((ext_vector_type(16)));

#define CIN   1024
#define HIDC  512
#define HW    1024          // 32*32
#define KTOT  (CIN * 9)     // 9216, reordered as K = r*CIN + cin
#define KT    32            // K per slab (one bf16 WMMA K-depth)
#define NSLAB (KTOT / KT)   // 288 = 9 taps * 32 slabs; slab never crosses a tap
#define BM    64
#define BN    128
#define NANC  30
#define AANC  (HW * NANC)   // 30720
#define NB    8
#define NGT   8

// ---------------------------------------------------------------------------
// fp32 -> (bf16 hi, bf16 lo): hi = truncated top 16 bits, lo = bf16 of the
// exact residual; hi + lo ~= f to ~2^-16 relative.
// ---------------------------------------------------------------------------
__device__ __forceinline__ void split2(float f, unsigned short& h, unsigned short& l)
{
    unsigned int u = __float_as_uint(f);
    h = (unsigned short)(u >> 16);
    float fh = __uint_as_float(u & 0xFFFF0000u);
    l = (unsigned short)(__float_as_uint(f - fh) >> 16);
}

// ---------------------------------------------------------------------------
// Pre-split kernel 1: weights fp32 [oc][cin][r] -> bf16 hi/lo [oc][r][cin]
// (tap-major K order so every K-slab of 32 has a constant 3x3 tap)
// ---------------------------------------------------------------------------
__global__ __launch_bounds__(256) void split_weights(
    const float* __restrict__ w, unsigned short* __restrict__ wH,
    unsigned short* __restrict__ wL)
{
    int tid = blockIdx.x * 256 + threadIdx.x;         // over 512*9216
    int oc  = tid / KTOT;
    int rem = tid - oc * KTOT;
    int r   = rem >> 10;                              // 0..8
    int cin = rem & 1023;
    float f = w[(size_t)oc * KTOT + cin * 9 + r];
    unsigned short h, l;
    split2(f, h, l);
    wH[tid] = h;
    wL[tid] = l;
}

// ---------------------------------------------------------------------------
// Pre-split kernel 2: fm fp32 [b][cin][pix] -> bf16 hi/lo [b][pix][cin]
// (transpose via LDS tile so the conv's im2col gather is cin-contiguous)
// ---------------------------------------------------------------------------
__global__ __launch_bounds__(256) void split_fm(
    const float* __restrict__ fm, unsigned short* __restrict__ fmH,
    unsigned short* __restrict__ fmL)
{
    __shared__ float tile[32][33];
    const int b    = blockIdx.z;
    const int cin0 = blockIdx.y * 32;
    const int pix0 = blockIdx.x * 32;
    const int tx   = threadIdx.x & 31;
    const int ty   = threadIdx.x >> 5;                // 32 x 8 threads
#pragma unroll
    for (int i = 0; i < 4; ++i) {
        int cin = cin0 + ty + i * 8;
        tile[ty + i * 8][tx] = fm[((size_t)b * CIN + cin) * HW + pix0 + tx];
    }
    __syncthreads();
#pragma unroll
    for (int i = 0; i < 4; ++i) {
        int pix = pix0 + ty + i * 8;
        float f = tile[tx][ty + i * 8];
        unsigned short h, l;
        split2(f, h, l);
        size_t o = ((size_t)b * HW + pix) * CIN + cin0 + tx;
        fmH[o] = h;
        fmL[o] = l;
    }
}

// ---------------------------------------------------------------------------
// Kernel 1: 3x3 conv as implicit GEMM, bf16x3 fp32 emulation on
// v_wmma_f32_16x16x32_bf16, staged from pre-split bf16 operands.
// Block 256 thr = 8 waves; block tile 64x128; wave tile 32x32 (2x2 WMMA x3).
// Register->LDS double-buffered K pipeline, one barrier per slab.
// ---------------------------------------------------------------------------
__global__ __launch_bounds__(256) void conv3x3_wmma_bf16x3(
    const unsigned short* __restrict__ wH,   // [HID][9][CIN] bf16 hi
    const unsigned short* __restrict__ wL,   // [HID][9][CIN] bf16 lo
    const unsigned short* __restrict__ fmH,  // [B][pix][CIN] bf16 hi
    const unsigned short* __restrict__ fmL,  // [B][pix][CIN] bf16 lo
    const float* __restrict__ bias,          // [HID]
    float* __restrict__ hout)                // [B, HID, 32, 32]
{
    // packed bf16-pair tiles (u32 = K,K+1), hi/lo, double buffered: 48KB
    __shared__ alignas(32) unsigned int AsH[2][BM * 16];  // [m][perm(k2)]
    __shared__ alignas(32) unsigned int AsL[2][BM * 16];
    __shared__ alignas(32) unsigned int BsH[2][BN * 16];  // [p][k2]
    __shared__ alignas(32) unsigned int BsL[2][BN * 16];

    const int t      = threadIdx.x;
    const int lane   = t & 31;
    const int wave   = t >> 5;
    const int b      = blockIdx.z;
    const int ocBase = blockIdx.y * BM;
    const int pBase  = blockIdx.x * BN;
    const int wm     = (wave & 1) * 32;
    const int wn     = (wave >> 1) * 32;

    v8f acc[2][2];
#pragma unroll
    for (int i = 0; i < 2; ++i)
#pragma unroll
        for (int j = 0; j < 2; ++j)
#pragma unroll
            for (int e = 0; e < 8; ++e) acc[i][j][e] = 0.0f;

    // static per-thread staging roles
    const int aM    = t >> 2;                         // A: oc row 0..63
    const int aQ    = t & 3;                          // A: quarter (4 pairs)
    const int aSb   = aQ << 2;                        // {0,4,8,12}: contiguous base
    const int bP    = t >> 1;                         // B: pixel 0..127
    const int bHalf = t & 1;                          // B: k2 half
    const int bPg   = pBase + bP;
    const int bY    = bPg >> 5;
    const int bX    = bPg & 31;

    const unsigned short* wHrow = wH + (size_t)(ocBase + aM) * KTOT;
    const unsigned short* wLrow = wL + (size_t)(ocBase + aM) * KTOT;
    const unsigned short* fHb   = fmH + (size_t)b * HW * CIN;
    const unsigned short* fLb   = fmL + (size_t)b * HW * CIN;

    // staging registers (next slab)
    uint4 aH4, aL4;
    v8u   bH8, bL8;

    auto loadRegs = [&](int slab) {
        const int r    = slab >> 5;                   // tap 0..8 (uniform)
        const int cin0 = (slab & 31) << 5;            // 32 cins per slab
        // A: 64 oc x 32 K bf16 -> thread loads 16B (4 K-pairs)
        {
            const size_t e = (size_t)r * CIN + cin0;  // even, 16B-aligned
            aH4 = ((const uint4*)(wHrow + e))[aQ];
            aL4 = ((const uint4*)(wLrow + e))[aQ];
        }
        // B: 128 pix x 32 K bf16 -> thread loads 32B (8 K-pairs, one pixel)
        {
            const int ky = r / 3;
            const int kx = r - ky * 3;
            const int iy = bY + ky - 1;
            const int ix = bX + kx - 1;
            if ((unsigned)iy < 32u && (unsigned)ix < 32u) {
                const size_t e = ((size_t)(iy * 32 + ix)) * CIN + cin0 + bHalf * 16;
                bH8 = *(const v8u*)(fHb + e);
                bL8 = *(const v8u*)(fLb + e);
            } else {
                const v8u z = {0, 0, 0, 0, 0, 0, 0, 0};
                bH8 = z;
                bL8 = z;
            }
        }
    };

    auto storeLds = [&](int buf) {
        *(uint4*)&AsH[buf][aM * 16 + aSb] = aH4;      // quarter -> contiguous perm
        *(uint4*)&AsL[buf][aM * 16 + aSb] = aL4;
        *(v8u*)&BsH[buf][bP * 16 + bHalf * 8] = bH8;
        *(v8u*)&BsL[buf][bP * 16 + bHalf * 8] = bL8;
    };

    auto compute = [&](int buf) {
        const int h  = lane >> 4;                     // lane half selects K block
        const int ln = lane & 15;
        v16bf aH[2], aL[2], bH[2], bL[2];
#pragma unroll
        for (int tM = 0; tM < 2; ++tM) {
            int m = wm + tM * 16 + ln;
            aH[tM] = __builtin_bit_cast(v16bf, *(const v8u*)&AsH[buf][m * 16 + h * 8]);
            aL[tM] = __builtin_bit_cast(v16bf, *(const v8u*)&AsL[buf][m * 16 + h * 8]);
        }
#pragma unroll
        for (int tN = 0; tN < 2; ++tN) {
            int p = wn + tN * 16 + ln;
            bH[tN] = __builtin_bit_cast(v16bf, *(const v8u*)&BsH[buf][p * 16 + h * 8]);
            bL[tN] = __builtin_bit_cast(v16bf, *(const v8u*)&BsL[buf][p * 16 + h * 8]);
        }
#pragma unroll
        for (int tM = 0; tM < 2; ++tM)
#pragma unroll
            for (int tN = 0; tN < 2; ++tN) {
                acc[tM][tN] = __builtin_amdgcn_wmma_f32_16x16x32_bf16(
                    false, aH[tM], false, bH[tN], (short)0, acc[tM][tN], false, false);
                acc[tM][tN] = __builtin_amdgcn_wmma_f32_16x16x32_bf16(
                    false, aH[tM], false, bL[tN], (short)0, acc[tM][tN], false, false);
                acc[tM][tN] = __builtin_amdgcn_wmma_f32_16x16x32_bf16(
                    false, aL[tM], false, bH[tN], (short)0, acc[tM][tN], false, false);
            }
    };

    loadRegs(0);
    storeLds(0);
    __syncthreads();

    for (int s = 0; s < NSLAB; ++s) {
        const int cur = s & 1;
        const bool more = (s + 1) < NSLAB;
        if (more) loadRegs(s + 1);     // global loads in flight over compute
        compute(cur);
        if (more) storeLds(cur ^ 1);   // waits loads, fills other buffer
        __syncthreads();
    }

    // Epilogue: bias + ReLU.  C/D layout: N = lane%16, M = v + 8*(lane/16)
    const int n   = lane & 15;
    const int mhi = (lane >> 4) * 8;
#pragma unroll
    for (int tM = 0; tM < 2; ++tM) {
#pragma unroll
        for (int tN = 0; tN < 2; ++tN) {
            int p = pBase + wn + tN * 16 + n;
#pragma unroll
            for (int v = 0; v < 8; ++v) {
                int oc = ocBase + wm + tM * 16 + mhi + v;
                float val = acc[tM][tN][v] + bias[oc];
                val = fmaxf(val, 0.0f);
                hout[(size_t)(b * HIDC + oc) * HW + p] = val;
            }
        }
    }
}

// ---------------------------------------------------------------------------
// Kernel 2: fused 1x1 conv heads (conf 30ch + reg 120ch), K = 512
// ---------------------------------------------------------------------------
__global__ __launch_bounds__(256) void heads_1x1(
    const float* __restrict__ hin,    // [B, HID, 32, 32]
    const float* __restrict__ wconf,  // [30, HID]
    const float* __restrict__ bconf,  // [30]
    const float* __restrict__ wreg,   // [120, HID]
    const float* __restrict__ breg,   // [120]
    float* __restrict__ conf_out,     // [B, A]
    float* __restrict__ reg_out)      // [B, A, 4]
{
    __shared__ float hT[HIDC * 16];   // [c][xi]  32 KB

    const int blk  = blockIdx.x;      // b*64 + y*2 + half
    const int half = blk & 1;
    const int y    = (blk >> 1) & 31;
    const int b    = blk >> 6;
    const int x0   = half * 16;
    const int t    = threadIdx.x;

    for (int i = t; i < HIDC * 16; i += 256) {
        int c  = i >> 4;
        int xi = i & 15;
        hT[c * 16 + xi] = hin[(size_t)(b * HIDC + c) * HW + y * 32 + x0 + xi];
    }
    __syncthreads();

    for (int i = t; i < 150 * 16; i += 256) {
        int oc = i >> 4;
        int xi = i & 15;
        const float* wv;
        float s;
        if (oc < 30) { wv = wconf + oc * HIDC; s = bconf[oc]; }
        else         { wv = wreg + (oc - 30) * HIDC; s = breg[oc - 30]; }
#pragma unroll 4
        for (int c = 0; c < HIDC; ++c) s += wv[c] * hT[c * 16 + xi];
        int p = y * 32 + x0 + xi;
        if (oc < 30) {
            conf_out[(size_t)b * AANC + p * NANC + oc] = s;
        } else {
            int ocr = oc - 30;
            int anc = ocr >> 2;
            int cd  = ocr & 3;
            reg_out[(((size_t)b * AANC + p * NANC + anc) << 2) + cd] = s;
        }
    }
}

// ---------------------------------------------------------------------------
// Anchor / IoU helpers (bit-identical across kernels for the tie-match rule)
// ---------------------------------------------------------------------------
__device__ __forceinline__ void anchor_wh(int anc, float& w, float& h)
{
    const float scales[5] = {0.25f, 0.5f, 0.75f, 1.0f, 2.0f};
    const float ratios[3] = {0.5f, 1.0f, 1.5f};
    if (anc < 15) { int s = anc / 3, r = anc - s * 3; w = scales[s] * ratios[r]; h = scales[s]; }
    else { int j = anc - 15; int s = j / 3, r = j - s * 3; w = scales[s]; h = scales[s] * ratios[r]; }
}

__device__ __forceinline__ float iou_box(float ax0, float ay0, float ax1, float ay1,
                                         float gx0, float gy0, float gx1, float gy1)
{
    float ltx = fmaxf(ax0, gx0), lty = fmaxf(ay0, gy0);
    float rbx = fminf(ax1, gx1), rby = fminf(ay1, gy1);
    float wx = fmaxf(rbx - ltx, 0.0f), wy = fmaxf(rby - lty, 0.0f);
    float inter = wx * wy;
    float aa = (ax1 - ax0) * (ay1 - ay0);
    float ag = (gx1 - gx0) * (gy1 - gy0);
    return inter / (aa + ag - inter + 1e-8f);
}

// ---------------------------------------------------------------------------
// Kernel 3: max IoU over anchors for each (b, gt)
// ---------------------------------------------------------------------------
__global__ __launch_bounds__(256) void max_iou_per_gt(
    const float* __restrict__ gtb,    // [B, G, 4] image coords
    float* __restrict__ maxgt)        // [B*G]
{
    __shared__ float red[256];
    const int bg = blockIdx.x;
    const int b = bg >> 3, g = bg & 7;
    const float inv = 1.0f / 16.0f;
    const float gx0 = gtb[(b * NGT + g) * 4 + 0] * inv;
    const float gy0 = gtb[(b * NGT + g) * 4 + 1] * inv;
    const float gx1 = gtb[(b * NGT + g) * 4 + 2] * inv;
    const float gy1 = gtb[(b * NGT + g) * 4 + 3] * inv;

    float m = 0.0f;
    for (int a = threadIdx.x; a < AANC; a += 256) {
        int cell = a / NANC, anc = a - cell * NANC;
        int y = cell >> 5, x = cell & 31;
        float w, h;
        anchor_wh(anc, w, h);
        float cx = x + 0.5f, cy = y + 0.5f;
        float iou = iou_box(cx - 0.5f * w, cy - 0.5f * h, cx + 0.5f * w, cy + 0.5f * h,
                            gx0, gy0, gx1, gy1);
        m = fmaxf(m, iou);
    }
    red[threadIdx.x] = m;
    __syncthreads();
    for (int s = 128; s > 0; s >>= 1) {
        if (threadIdx.x < s) red[threadIdx.x] = fmaxf(red[threadIdx.x], red[threadIdx.x + s]);
        __syncthreads();
    }
    if (threadIdx.x == 0) maxgt[bg] = red[0];
}

// ---------------------------------------------------------------------------
// Kernel 4: anchor assignment, losses (partial sums), proposals, pos_mask
// ---------------------------------------------------------------------------
__global__ __launch_bounds__(256) void rpn_assign(
    const float* __restrict__ gtb,
    const float* __restrict__ maxgt,
    const float* __restrict__ conf_ws,   // [B, A]
    const float* __restrict__ reg_ws,    // [B, A, 4]
    float* __restrict__ out_prop,        // [B, A, 4]
    float* __restrict__ out_mask,        // [B, A]
    float* __restrict__ partial)         // [gridDim.x, 2]
{
    const int idx = blockIdx.x * 256 + threadIdx.x;  // b*A + a
    const int b = idx / AANC;
    const int a = idx - b * AANC;
    const int cell = a / NANC, anc = a - cell * NANC;
    const int y = cell >> 5, x = cell & 31;

    float aw, ah;
    anchor_wh(anc, aw, ah);
    const float acx = x + 0.5f, acy = y + 0.5f;
    const float ax0 = acx - 0.5f * aw, ay0 = acy - 0.5f * ah;
    const float ax1 = acx + 0.5f * aw, ay1 = acy + 0.5f * ah;

    const float inv = 1.0f / 16.0f;
    float best = -1.0f;
    int bestG = 0;
    bool pos = false;
    for (int g = 0; g < NGT; ++g) {
        float gx0 = gtb[(b * NGT + g) * 4 + 0] * inv;
        float gy0 = gtb[(b * NGT + g) * 4 + 1] * inv;
        float gx1 = gtb[(b * NGT + g) * 4 + 2] * inv;
        float gy1 = gtb[(b * NGT + g) * 4 + 3] * inv;
        float iou = iou_box(ax0, ay0, ax1, ay1, gx0, gy0, gx1, gy1);
        if (iou > best) { best = iou; bestG = g; }     // first-argmax semantics
        float mg = maxgt[b * NGT + g];
        pos = pos || (iou > 0.7f) || ((iou == mg) && (mg > 0.0f));
    }
    const bool neg = (best < 0.3f) && !pos;

    float gx0 = gtb[(b * NGT + bestG) * 4 + 0] * inv;
    float gy0 = gtb[(b * NGT + bestG) * 4 + 1] * inv;
    float gx1 = gtb[(b * NGT + bestG) * 4 + 2] * inv;
    float gy1 = gtb[(b * NGT + bestG) * 4 + 3] * inv;
    float gw = gx1 - gx0, gh = gy1 - gy0;
    float gcx = gx0 + 0.5f * gw, gcy = gy0 + 0.5f * gh;
    float off0 = (gcx - acx) / aw;
    float off1 = (gcy - acy) / ah;
    float off2 = logf(fmaxf(gw, 1e-6f) / aw);
    float off3 = logf(fmaxf(gh, 1e-6f) / ah);

    float cl = conf_ws[idx];
    float r0 = reg_ws[idx * 4 + 0];
    float r1 = reg_ws[idx * 4 + 1];
    float r2 = reg_ws[idx * 4 + 2];
    float r3 = reg_ws[idx * 4 + 3];

    const float tgt = pos ? 1.0f : 0.0f;
    float bce = fmaxf(cl, 0.0f) - cl * tgt + log1pf(expf(-fabsf(cl)));
    float clsC = (pos || neg) ? bce : 0.0f;

    float regC = 0.0f;
    if (pos) {
        float d, s = 0.0f;
        d = fabsf(r0 - off0); s += (d < 1.0f) ? 0.5f * d * d : (d - 0.5f);
        d = fabsf(r1 - off1); s += (d < 1.0f) ? 0.5f * d * d : (d - 0.5f);
        d = fabsf(r2 - off2); s += (d < 1.0f) ? 0.5f * d * d : (d - 0.5f);
        d = fabsf(r3 - off3); s += (d < 1.0f) ? 0.5f * d * d : (d - 0.5f);
        regC = s;
    }

    float pcx = acx + r0 * aw;
    float pcy = acy + r1 * ah;
    float pw = aw * expf(r2);
    float ph = ah * expf(r3);
    out_prop[idx * 4 + 0] = pcx - 0.5f * pw;
    out_prop[idx * 4 + 1] = pcy - 0.5f * ph;
    out_prop[idx * 4 + 2] = pcx + 0.5f * pw;
    out_prop[idx * 4 + 3] = pcy + 0.5f * ph;
    out_mask[idx] = tgt;

    __shared__ float sc[256], sr[256];
    sc[threadIdx.x] = clsC;
    sr[threadIdx.x] = regC;
    __syncthreads();
    for (int s = 128; s > 0; s >>= 1) {
        if (threadIdx.x < s) {
            sc[threadIdx.x] += sc[threadIdx.x + s];
            sr[threadIdx.x] += sr[threadIdx.x + s];
        }
        __syncthreads();
    }
    if (threadIdx.x == 0) {
        partial[blockIdx.x * 2 + 0] = sc[0];
        partial[blockIdx.x * 2 + 1] = sr[0];
    }
}

// ---------------------------------------------------------------------------
// Kernel 5: deterministic final reduction -> total loss scalar
// ---------------------------------------------------------------------------
__global__ __launch_bounds__(256) void finalize_loss(
    const float* __restrict__ partial, int nblk, float* __restrict__ out_total)
{
    __shared__ float sc[256], sr[256];
    float c = 0.0f, r = 0.0f;
    for (int i = threadIdx.x; i < nblk; i += 256) {
        c += partial[i * 2 + 0];
        r += partial[i * 2 + 1];
    }
    sc[threadIdx.x] = c;
    sr[threadIdx.x] = r;
    __syncthreads();
    for (int s = 128; s > 0; s >>= 1) {
        if (threadIdx.x < s) {
            sc[threadIdx.x] += sc[threadIdx.x + s];
            sr[threadIdx.x] += sr[threadIdx.x + s];
        }
        __syncthreads();
    }
    if (threadIdx.x == 0)
        out_total[0] = (sc[0] + 5.0f * sr[0]) * (1.0f / (float)NB);
}

// ---------------------------------------------------------------------------
extern "C" void kernel_launch(void* const* d_in, const int* in_sizes, int n_in,
                              void* d_out, int out_size, void* d_ws, size_t ws_size,
                              hipStream_t stream)
{
    const float* fm     = (const float*)d_in[0];
    const float* gtb    = (const float*)d_in[1];
    // d_in[2] = gt_classes (unused by reference outputs)
    const float* w_conv = (const float*)d_in[3];
    const float* b_conv = (const float*)d_in[4];
    const float* w_conf = (const float*)d_in[5];
    const float* b_conf = (const float*)d_in[6];
    const float* w_reg  = (const float*)d_in[7];
    const float* b_reg  = (const float*)d_in[8];

    char* base = (char*)d_ws;
    size_t off = 0;
    float* h = (float*)(base + off);          off += (size_t)NB * HIDC * HW * 4;     // 16.78 MB
    float* conf = (float*)(base + off);       off += (size_t)NB * AANC * 4;          // 0.98 MB
    float* reg = (float*)(base + off);        off += (size_t)NB * AANC * 4 * 4;      // 3.93 MB
    float* maxgt = (float*)(base + off);      off += (size_t)NB * NGT * 4;           // 256 B
    float* partial = (float*)(base + off);    off += (size_t)2 * ((NB * AANC) / 256) * 4;
    off = (off + 255) & ~(size_t)255;
    unsigned short* wH = (unsigned short*)(base + off);  off += (size_t)HIDC * KTOT * 2;  // 9.44 MB
    unsigned short* wL = (unsigned short*)(base + off);  off += (size_t)HIDC * KTOT * 2;
    unsigned short* fH = (unsigned short*)(base + off);  off += (size_t)NB * HW * CIN * 2; // 16.78 MB
    unsigned short* fL = (unsigned short*)(base + off);  off += (size_t)NB * HW * CIN * 2;

    float* out_total = (float*)d_out;
    float* out_prop  = out_total + 1;
    float* out_mask  = out_prop + (size_t)NB * AANC * 4;

    split_weights<<<(HIDC * KTOT) / 256, 256, 0, stream>>>(w_conv, wH, wL);
    split_fm<<<dim3(HW / 32, CIN / 32, NB), 256, 0, stream>>>(fm, fH, fL);

    dim3 cgrid(HW / BN, HIDC / BM, NB);             // (8, 8, 8)
    conv3x3_wmma_bf16x3<<<cgrid, 256, 0, stream>>>(wH, wL, fH, fL, b_conv, h);
    heads_1x1<<<NB * 64, 256, 0, stream>>>(h, w_conf, b_conf, w_reg, b_reg, conf, reg);
    max_iou_per_gt<<<NB * NGT, 256, 0, stream>>>(gtb, maxgt);
    const int nblk = (NB * AANC) / 256;             // 960
    rpn_assign<<<nblk, 256, 0, stream>>>(gtb, maxgt, conf, reg, out_prop, out_mask, partial);
    finalize_loss<<<1, 256, 0, stream>>>(partial, nblk, out_total);
}
